// ClusterMemory_15710990369516
// MI455X (gfx1250) — matrix operand order repaired and checked by reading, not compiled, not run on hardware
//
#include <hip/hip_runtime.h>
#include <hip/hip_bf16.h>

// Problem sizes (fixed by the reference)
#define BB   256      // batch
#define DD   2048     // feature dim
#define NN   16384    // memory bank rows
#define TEMP_INV 20.0f

// GEMM tiling
#define KT     32                 // K per step (bf16 WMMA K)
#define MBLK   64                 // rows per block
#define NBLK   128                // cols per block (8 waves x 16)
#define NCHUNK (NN / NBLK)        // 128 column chunks
#define KP     40                 // padded LDS K stride (halves)
#define NSTEPS (DD / KT)          // 64 K-steps

typedef __attribute__((ext_vector_type(16))) __bf16 v16bf;
typedef __attribute__((ext_vector_type(16), aligned(16))) __bf16 v16bf_a16;
typedef __attribute__((ext_vector_type(8)))  __bf16 bf16x8;
typedef __attribute__((ext_vector_type(8)))  float  v8f;

// 16-lane XOR reductions via ds_swizzle group-of-32 patterns (wave32).
// Offset encoding: and_mask | (or_mask<<5) | (xor_mask<<10); SWAPXn = xor n.
#define SWZ_X1 0x041F
#define SWZ_X2 0x081F
#define SWZ_X4 0x101F
#define SWZ_X8 0x201F
template <int IMM>
__device__ __forceinline__ float swz(float x) {
  return __int_as_float(__builtin_amdgcn_ds_swizzle(__float_as_int(x), IMM));
}

// ---------------------------------------------------------------------------
// Kernel 1: L2-normalize inputs rows, emit bf16 copy for the WMMA A operand.
// ---------------------------------------------------------------------------
__global__ __launch_bounds__(256) void normalize_kernel(
    const float* __restrict__ in, __bf16* __restrict__ xb)
{
  __shared__ float red[256];
  const int b = blockIdx.x, tid = threadIdx.x;
  float acc = 0.f;
  for (int d = tid; d < DD; d += 256) { float v = in[b * DD + d]; acc += v * v; }
  red[tid] = acc; __syncthreads();
  for (int s = 128; s > 0; s >>= 1) {
    if (tid < s) red[tid] += red[tid + s];
    __syncthreads();
  }
  const float inv = rsqrtf(red[0]);
  for (int d = tid; d < DD; d += 256)
    xb[b * DD + d] = (__bf16)(in[b * DD + d] * inv);
}

// ---------------------------------------------------------------------------
// Kernel 2: sims = (x @ features^T)/TEMP via bf16 WMMA, fused online softmax
// partials. Block: 64 rows x 128 cols. Double-buffered LDS, register-staged
// global prefetch, one barrier per K-step, batched frag loads -> 4 WMMAs.
// ---------------------------------------------------------------------------
__global__ __launch_bounds__(256) void gemm_softmax_partial(
    const __bf16* __restrict__ xb,     // [BB][DD] normalized, bf16
    const float*  __restrict__ feats,  // [NN][DD] f32 (L2-normalized rows)
    float* __restrict__ pm, float* __restrict__ ps)
{
  __shared__ alignas(16) __bf16 Apan[2][MBLK * KP];
  __shared__ alignas(16) __bf16 Bpan[2][NBLK * KP];
  __shared__ float redm[MBLK * 8];
  __shared__ float reds[MBLK * 8];

  const int tid  = threadIdx.x;
  const int wave = tid >> 5;
  const int lane = tid & 31;
  const int nchunk = blockIdx.x;      // 0..NCHUNK-1
  const int m0 = blockIdx.y * MBLK;   // 0,64,128,192
  const int n0 = nchunk * NBLK;

  v8f acc[4];
  const v8f vzero = {0.f,0.f,0.f,0.f,0.f,0.f,0.f,0.f};
#pragma unroll
  for (int i = 0; i < 4; ++i) acc[i] = vzero;

  // staging maps: A: 64 rows x 4 chunks of 8 halves; B: 128 rows x 2 x 16 f32
  const int arow = tid >> 2, achk = tid & 3;
  const int brow = tid >> 1, bhalf = tid & 1;
  const __bf16* agsrc = xb + (size_t)(m0 + arow) * DD + achk * 8;
  const float*  bgsrc = feats + (size_t)(n0 + brow) * DD + bhalf * 16;

  // fragment addresses (per lane)
  const int bcol  = (wave << 4) + (lane & 15);
  const int ksel  = (lane >> 4) << 4;  // 0 or 16 (B: K contiguous per half)
  const int abase = (lane >> 4) << 4;  // 0 or 16 (A: K contiguous per half)
  const int amrow = lane & 15;

  // register staging (data for step 'it' loaded during step 'it-1')
  bf16x8 aS;
  float4 bS0, bS1, bS2, bS3;
  {
    aS = *(const bf16x8*)(agsrc);
    const float4* src = (const float4*)(bgsrc);
    bS0 = src[0]; bS1 = src[1]; bS2 = src[2]; bS3 = src[3];
  }

  for (int it = 0; it < NSTEPS; ++it) {
    const int buf = it & 1;
    // ---- store staged registers into this step's LDS panels ----
    *(bf16x8*)(&Apan[buf][arow * KP + achk * 8]) = aS;
    {
      __bf16* dst = &Bpan[buf][brow * KP + bhalf * 16];
      dst[0]  = (__bf16)bS0.x; dst[1]  = (__bf16)bS0.y;
      dst[2]  = (__bf16)bS0.z; dst[3]  = (__bf16)bS0.w;
      dst[4]  = (__bf16)bS1.x; dst[5]  = (__bf16)bS1.y;
      dst[6]  = (__bf16)bS1.z; dst[7]  = (__bf16)bS1.w;
      dst[8]  = (__bf16)bS2.x; dst[9]  = (__bf16)bS2.y;
      dst[10] = (__bf16)bS2.z; dst[11] = (__bf16)bS2.w;
      dst[12] = (__bf16)bS3.x; dst[13] = (__bf16)bS3.y;
      dst[14] = (__bf16)bS3.z; dst[15] = (__bf16)bS3.w;
    }
    __syncthreads();

    // ---- issue next step's global loads (consumed at top of next iter) ----
    if (it + 1 < NSTEPS) {
      const int kn = (it + 1) * KT;
      aS = *(const bf16x8*)(agsrc + kn);
      const float4* src = (const float4*)(bgsrc + kn);
      bS0 = src[0]; bS1 = src[1]; bS2 = src[2]; bS3 = src[3];
    }
    // deep prefetch of the features stream (global_prefetch_b8)
    if (it + 4 < NSTEPS)
      __builtin_prefetch((const void*)(bgsrc + (it + 4) * KT), 0, 3);

    // ---- batch ALL fragment loads, then chain 4 WMMAs ----
    const v16bf bfrag = *(const v16bf_a16*)(&Bpan[buf][bcol * KP + ksel]);
    bf16x8 alo[4], ahi[4];
#pragma unroll
    for (int ms = 0; ms < 4; ++ms) {
      const __bf16* ap = &Apan[buf][((ms << 4) + amrow) * KP + abase];
      alo[ms] = *(const bf16x8*)(ap);
      ahi[ms] = *(const bf16x8*)(ap + 8);
    }
#pragma unroll
    for (int ms = 0; ms < 4; ++ms) {
      v16bf afrag;
#pragma unroll
      for (int i = 0; i < 8; ++i) { afrag[i] = alo[ms][i]; afrag[i + 8] = ahi[ms][i]; }
      acc[ms] = __builtin_amdgcn_wmma_f32_16x16x32_bf16(
          false, afrag, false, bfrag, (short)0, acc[ms], false, false);
    }
    // no trailing barrier: next iteration writes the *other* LDS buffer
  }

  // ---- epilogue: scale by 1/TEMP; per-row (max, sumexp) over 16 cols ----
#pragma unroll
  for (int ms = 0; ms < 4; ++ms) {
#pragma unroll
    for (int v = 0; v < 8; ++v) {
      const float val = acc[ms][v] * TEMP_INV;
      float rm = val;
      rm = fmaxf(rm, swz<SWZ_X1>(rm));
      rm = fmaxf(rm, swz<SWZ_X2>(rm));
      rm = fmaxf(rm, swz<SWZ_X4>(rm));
      rm = fmaxf(rm, swz<SWZ_X8>(rm));
      float e = __expf(val - rm);
      e += swz<SWZ_X1>(e);
      e += swz<SWZ_X2>(e);
      e += swz<SWZ_X4>(e);
      e += swz<SWZ_X8>(e);
      if ((lane & 15) == 0) {
        const int row = (ms << 4) + v + ((lane >> 4) << 3);
        redm[row * 8 + wave] = rm;
        reds[row * 8 + wave] = e;
      }
    }
  }
  __syncthreads();

  if (tid < MBLK) {
    float M = -3.4e38f, S = 0.f;
#pragma unroll
    for (int w = 0; w < 8; ++w) M = fmaxf(M, redm[tid * 8 + w]);
#pragma unroll
    for (int w = 0; w < 8; ++w) S += reds[tid * 8 + w] * __expf(redm[tid * 8 + w] - M);
    const int grow = m0 + tid;
    pm[grow * NCHUNK + nchunk] = M;
    ps[grow * NCHUNK + nchunk] = S;
  }
}

// ---------------------------------------------------------------------------
// Kernel 3: merge 128 (max,sum) partials per row -> logZ[row]
// ---------------------------------------------------------------------------
__global__ __launch_bounds__(128) void logz_kernel(
    const float* __restrict__ pm, const float* __restrict__ ps,
    float* __restrict__ logZ)
{
  __shared__ float rm[128], rs[128];
  const int row = blockIdx.x, tid = threadIdx.x;
  const float m = pm[row * NCHUNK + tid];
  const float s = ps[row * NCHUNK + tid];
  rm[tid] = m; __syncthreads();
  for (int st = 64; st > 0; st >>= 1) {
    if (tid < st) rm[tid] = fmaxf(rm[tid], rm[tid + st]);
    __syncthreads();
  }
  const float M = rm[0];
  rs[tid] = s * __expf(m - M); __syncthreads();
  for (int st = 64; st > 0; st >>= 1) {
    if (tid < st) rs[tid] += rs[tid + st];
    __syncthreads();
  }
  if (tid == 0) logZ[row] = M + logf(rs[0]);
}

// ---------------------------------------------------------------------------
// Kernel 4: s_own[i] = dot(normalize(inputs[i]), features[targets[i]]) / TEMP
// ---------------------------------------------------------------------------
__global__ __launch_bounds__(256) void sown_kernel(
    const float* __restrict__ in, const float* __restrict__ feats,
    const int* __restrict__ targets, float* __restrict__ s_own)
{
  __shared__ float r1[256], r2[256];
  const int i = blockIdx.x, tid = threadIdx.x;
  const int t = targets[i];
  const float* xr = in + (size_t)i * DD;
  const float* fr = feats + (size_t)t * DD;
  float a1 = 0.f, a2 = 0.f;
  for (int d = tid; d < DD; d += 256) {
    const float v = xr[d];
    a1 += v * fr[d];
    a2 += v * v;
  }
  r1[tid] = a1; r2[tid] = a2; __syncthreads();
  for (int s = 128; s > 0; s >>= 1) {
    if (tid < s) { r1[tid] += r1[tid + s]; r2[tid] += r2[tid + s]; }
    __syncthreads();
  }
  if (tid == 0) s_own[i] = r1[0] * rsqrtf(r2[0]) * TEMP_INV;
}

// ---------------------------------------------------------------------------
// Kernel 5: B x B group logic + final loss (single block of 256 threads)
// ---------------------------------------------------------------------------
__global__ __launch_bounds__(256) void loss_kernel(
    const int* __restrict__ targets, const int* __restrict__ cams,
    const float* __restrict__ s_own, const float* __restrict__ logZ,
    float* __restrict__ out)
{
  __shared__ int   ts[BB];
  __shared__ int   cs[BB];
  __shared__ float so[BB];
  __shared__ float gf[BB];
  __shared__ float red[BB], red2[BB];
  const int i = threadIdx.x;
  ts[i] = targets[i]; cs[i] = cams[i]; so[i] = s_own[i];
  __syncthreads();
  const int ti = ts[i], ci = cs[i];
  const float si = so[i];

  float gmin = 3.4e38f;
  for (int j = 0; j < BB; ++j)
    if (ts[j] == ti && cs[j] == ci) gmin = fminf(gmin, so[j]);
  const bool is_min = (si <= gmin);

  bool any_psid = false, any_grp = false, any_min = false;
  for (int j = 0; j < i; ++j) {
    const bool sp = (ts[j] == ti);
    const bool sg = sp && (cs[j] == ci);
    any_psid |= sp;
    any_grp  |= sg;
    any_min  |= (sg && (so[j] <= gmin));   // gmin_j == gmin for same group
  }
  const bool hard_rep = is_min && !any_min;
  gf[i] = any_grp ? 0.f : 1.f;             // grp_first
  __syncthreads();

  float ng = 0.f;
  for (int j = 0; j < BB; ++j)
    if (ts[j] == ti) ng += gf[j];

  const float per = logZ[i] - si;          // -log_softmax at target column
  red[i]  = hard_rep ? (per / ng) : 0.f;
  red2[i] = any_psid ? 0.f : 1.f;          // psid_first
  __syncthreads();
  for (int s = 128; s > 0; s >>= 1) {
    if (i < s) { red[i] += red[i + s]; red2[i] += red2[i + s]; }
    __syncthreads();
  }
  if (i == 0) out[0] = red[0] / red2[0];
}

// ---------------------------------------------------------------------------
extern "C" void kernel_launch(void* const* d_in, const int* in_sizes, int n_in,
                              void* d_out, int out_size, void* d_ws, size_t ws_size,
                              hipStream_t stream) {
  const float* inputs  = (const float*)d_in[0];   // [256,2048]
  const float* feats   = (const float*)d_in[1];   // [16384,2048]
  const int*   targets = (const int*)d_in[2];     // [256]
  const int*   cams    = (const int*)d_in[3];     // [256]
  float* out = (float*)d_out;

  // workspace layout (~1.3 MB total)
  char* ws = (char*)d_ws;
  __bf16* xb  = (__bf16*)(ws);                            // 256*2048*2 = 1 MB
  float*  pm  = (float*)(ws + (1 << 20));                 // 256*128*4 = 128 KB
  float*  ps  = (float*)(ws + (1 << 20) + (128 << 10));   // 128 KB
  float*  lz  = (float*)(ws + (1 << 20) + (256 << 10));   // 1 KB
  float*  so  = (float*)(ws + (1 << 20) + (256 << 10) + 1024);

  normalize_kernel<<<BB, 256, 0, stream>>>(inputs, xb);
  dim3 grid(NCHUNK, BB / MBLK);                            // 128 x 4
  gemm_softmax_partial<<<grid, 256, 0, stream>>>(xb, feats, pm, ps);
  logz_kernel<<<BB, 128, 0, stream>>>(pm, ps, lz);
  sown_kernel<<<BB, 256, 0, stream>>>(inputs, feats, targets, so);
  loss_kernel<<<1, 256, 0, stream>>>(targets, cams, so, lz, out);
}
